// MultiHeadAttention_55070070670269
// MI455X (gfx1250) — matrix operand order repaired and checked by reading, not compiled, not run on hardware
//
#include <hip/hip_runtime.h>
#include <stdint.h>

// ---------------------------------------------------------------------------
// MultiHeadAttention (add_bias_kv) forward for MI455X (gfx1250, wave32, WMMA)
// L=2048, N=4, E=512, H=8, D=64, S=L+1=2049 (padded to 2080)
// All matmuls via v_wmma_f32_16x16x32_bf16 (f32 accumulate).
// All f32->bf16 conversion hoisted out of GEMM inner loops.
// ---------------------------------------------------------------------------

typedef __attribute__((ext_vector_type(16))) __bf16 v16bf;
typedef __attribute__((ext_vector_type(8)))  float  v8f;

#define LSEQ 2048
#define NB   4
#define EMB  512
#define NH   8
#define HD   64
#define BH   32      // NB*NH batched heads
#define SEQS 2049    // L + 1 bias token
#define SPAD 2080    // padded to multiple of 32 key-tiles
#define E3   1536

// ---- bf16 helpers (round-to-nearest-even, software; only used off hot path)
__device__ __forceinline__ uint16_t f2bf(float f) {
  uint32_t u = __builtin_bit_cast(uint32_t, f);
  uint32_t r = u + 0x7FFFu + ((u >> 16) & 1u);
  return (uint16_t)(r >> 16);
}

// ---- WMMA fragment loaders (contiguous k = 16h + e in both A and B, not the ISA table order) -----
// A (16x32 bf16, M x K): lane m = lane&15; per-lane element e holds
//   K = e + (lane>=16 ? 16 : 0)
__device__ __forceinline__ v16bf a_frag_bf(const uint16_t* row0, int stride, int lane) {
  const int half = (lane >> 4) & 1;
  const uint16_t* rp = row0 + (size_t)(lane & 15) * stride;
  v16bf a;
#pragma unroll
  for (int e = 0; e < 16; ++e) {
    int k = e + half * 16;
    a[e] = __builtin_bit_cast(__bf16, rp[k]);
  }
  return a;
}
// B (32x16 bf16, K x N): lanes 0-15 hold K=0..15 (N=lane), lanes 16-31 K=16..31.
// Memory row-major [N x K] (i.e. B = Matᵀ) -> 32 contiguous bytes per lane.
__device__ __forceinline__ v16bf b_frag_nk_bf(const uint16_t* m, int stride, int kbase, int lane) {
  const int half = (lane >> 4) & 1;
  const uint16_t* rp = m + (size_t)(lane & 15) * stride + kbase + half * 16;
  v16bf b;
#pragma unroll
  for (int e = 0; e < 16; ++e) b[e] = __builtin_bit_cast(__bf16, rp[e]);
  return b;
}

__device__ __forceinline__ v8f wmma_bf16(v16bf a, v16bf b, v8f c) {
  return __builtin_amdgcn_wmma_f32_16x16x32_bf16(false, a, false, b, (short)0, c, false, false);
}

// half-wave (16-lane) butterfly reductions; masks <=8 stay within a half.
__device__ __forceinline__ float half_reduce_max(float v) {
#pragma unroll
  for (int msk = 1; msk <= 8; msk <<= 1) v = fmaxf(v, __shfl_xor(v, msk, 32));
  return v;
}
__device__ __forceinline__ float half_reduce_sum(float v) {
#pragma unroll
  for (int msk = 1; msk <= 8; msk <<= 1) v += __shfl_xor(v, msk, 32);
  return v;
}

// ---------------------------------------------------------------------------
// Kernel 0a: elementwise f32 -> bf16 pre-conversion (memory bound, ~1us).
// ---------------------------------------------------------------------------
__global__ void k_cvt(const float* __restrict__ src, uint16_t* __restrict__ dst, int n) {
  int i = blockIdx.x * blockDim.x + threadIdx.x;
  if (i < n) dst[i] = f2bf(src[i]);
}

// ---------------------------------------------------------------------------
// Kernel 0b: zero K/V padding rows, write bias_k/bias_v token at s=2048.
// V is stored d-major: vt[bh][d][s] so the P@V B-fragment loads contiguously.
// ---------------------------------------------------------------------------
__global__ void k_init(const float* __restrict__ bias_k, const float* __restrict__ bias_v,
                       uint16_t* __restrict__ kb, uint16_t* __restrict__ vt) {
  int idx = blockIdx.x * blockDim.x + threadIdx.x;   // 32 heads * 32 rows * 64 d
  int d = idx & 63;
  int t = idx >> 6;
  int srow = LSEQ + (t & 31);                        // 2048..2079
  int b = t >> 5;                                    // 0..31
  int h = b & (NH - 1);
  float kv = (srow == LSEQ) ? bias_k[h * HD + d] : 0.f;
  float vv = (srow == LSEQ) ? bias_v[h * HD + d] : 0.f;
  kb[((size_t)b * SPAD + srow) * HD + d] = f2bf(kv);
  vt[((size_t)b * HD + d) * SPAD + srow] = f2bf(vv);
}

// ---------------------------------------------------------------------------
// Kernel 1: fused in_proj GEMM: (8192x512) @ Wᵀ(512x1536) + bias,
// split into per-head q (scaled), k, v-transposed bf16 buffers.
// ---------------------------------------------------------------------------
__global__ void __launch_bounds__(128)
k_qkv(const uint16_t* __restrict__ X, const uint16_t* __restrict__ W,
      const float* __restrict__ bias,
      uint16_t* __restrict__ qb, uint16_t* __restrict__ kb, uint16_t* __restrict__ vt) {
  const int lane = threadIdx.x & 31;
  const int wave = threadIdx.x >> 5;
  const int mt = blockIdx.x;                   // 0..511
  const int nt = blockIdx.y * 4 + wave;        // 0..95
  const int row0 = mt * 16;
  const int col  = nt * 16 + (lane & 15);
  const int half = (lane >> 4) & 1;

  v8f c = {};
#pragma unroll 4
  for (int kc = 0; kc < EMB; kc += 32) {
    __builtin_prefetch(X + (size_t)row0 * EMB + kc + 128, 0, 0);
    v16bf a = a_frag_bf(X + (size_t)row0 * EMB + kc, EMB, lane);
    v16bf b = b_frag_nk_bf(W + (size_t)nt * 16 * EMB, EMB, kc, lane);
    c = wmma_bf16(a, b, c);
  }
  const float bc = bias[col];
  const float scaling = 0.125f;                // D^-0.5 = 64^-0.5
#pragma unroll
  for (int r = 0; r < 8; ++r) {
    int rg = row0 + r + half * 8;              // global row = l*N + n
    int l = rg >> 2, n = rg & 3;
    float v = c[r] + bc;
    if (col < EMB) {                           // q
      int h = col >> 6, d = col & 63;
      qb[(((size_t)(n * NH + h)) * LSEQ + l) * HD + d] = f2bf(v * scaling);
    } else if (col < 2 * EMB) {                // k
      int e = col - EMB; int h = e >> 6, d = e & 63;
      kb[(((size_t)(n * NH + h)) * SPAD + l) * HD + d] = f2bf(v);
    } else {                                   // v (transposed: d-major)
      int e = col - 2 * EMB; int h = e >> 6, d = e & 63;
      vt[(((size_t)(n * NH + h)) * HD + d) * SPAD + l] = f2bf(v);
    }
  }
}

// ---------------------------------------------------------------------------
// Kernel 2: attention, exact 2-sweep softmax. 1 wave per (head, 16-query tile).
// Sweep 1: QKᵀ, per-lane column-subset max (no shuffles in the loop).
// Sweep 2: QKᵀ again, e = exp(score-max) once per score (masked), per-lane
//          row-sum partials, unnormalized P̃@V via WMMA (LDS C->A relayout),
//          then one cross-lane reduce + 1/denom normalization at the end.
// ---------------------------------------------------------------------------
__global__ void __launch_bounds__(32)
k_attn(const uint16_t* __restrict__ qb, const uint16_t* __restrict__ kb,
       const uint16_t* __restrict__ vt, uint16_t* __restrict__ ctx,
       float* __restrict__ rowmax, float* __restrict__ denom) {
  __shared__ uint16_t pstage[16 * 32];
  const int lane = threadIdx.x & 31;
  const int half = (lane >> 4) & 1;
  const int lt = blockIdx.x;                   // 0..127
  const int bh = blockIdx.y;                   // 0..31
  const int l0 = lt * 16;
  const uint16_t* qbase = qb + ((size_t)bh * LSEQ + l0) * HD;
  const uint16_t* kbase = kb + (size_t)bh * SPAD * HD;
  const uint16_t* vbase = vt + (size_t)bh * HD * SPAD;

  const v16bf aq0 = a_frag_bf(qbase,      HD, lane);   // K = d 0..31
  const v16bf aq1 = a_frag_bf(qbase + 32, HD, lane);   // K = d 32..63

  // ---- sweep 1: per-lane max over this lane's column subset
  float ml[8];
#pragma unroll
  for (int r = 0; r < 8; ++r) ml[r] = -1e30f;
  for (int st = 0; st < 129; ++st) {
    __builtin_prefetch(kbase + (size_t)(st + 2) * 16 * HD, 0, 0);
    v8f c = {};
    c = wmma_bf16(aq0, b_frag_nk_bf(kbase + (size_t)st * 16 * HD, HD, 0,  lane), c);
    c = wmma_bf16(aq1, b_frag_nk_bf(kbase + (size_t)st * 16 * HD, HD, 32, lane), c);
    const bool valid = (st * 16 + (lane & 15)) < SEQS;
#pragma unroll
    for (int r = 0; r < 8; ++r) ml[r] = fmaxf(ml[r], valid ? c[r] : -1e30f);
  }
#pragma unroll
  for (int r = 0; r < 8; ++r) ml[r] = half_reduce_max(ml[r]);  // final row max

  // ---- sweep 2: unnormalized probabilities + P̃@V + per-lane row sums
  float sl[8];
#pragma unroll
  for (int r = 0; r < 8; ++r) sl[r] = 0.f;
  v8f acc[4] = {};
  for (int sck = 0; sck < 65; ++sck) {
#pragma unroll
    for (int t = 0; t < 2; ++t) {
      const int st = sck * 2 + t;
      v8f c = {};
      c = wmma_bf16(aq0, b_frag_nk_bf(kbase + (size_t)st * 16 * HD, HD, 0,  lane), c);
      c = wmma_bf16(aq1, b_frag_nk_bf(kbase + (size_t)st * 16 * HD, HD, 32, lane), c);
      const bool valid = (st * 16 + (lane & 15)) < SEQS;
#pragma unroll
      for (int r = 0; r < 8; ++r) {
        float e = valid ? __expf(c[r] - ml[r]) : 0.f;
        sl[r] += e;
        pstage[(r + half * 8) * 32 + t * 16 + (lane & 15)] = f2bf(e);
      }
    }
    __syncthreads();
    v16bf ap = a_frag_bf(pstage, 32, lane);
#pragma unroll
    for (int nt4 = 0; nt4 < 4; ++nt4) {
      acc[nt4] = wmma_bf16(ap,
          b_frag_nk_bf(vbase + (size_t)nt4 * 16 * SPAD, SPAD, sck * 32, lane),
          acc[nt4]);
    }
    __syncthreads();
  }
#pragma unroll
  for (int r = 0; r < 8; ++r) sl[r] = half_reduce_sum(sl[r]);  // row denominators

  if ((lane & 15) == 0) {
#pragma unroll
    for (int r = 0; r < 8; ++r) {
      int row = l0 + r + half * 8;
      rowmax[(size_t)bh * LSEQ + row] = ml[r];
      denom [(size_t)bh * LSEQ + row] = sl[r];
    }
  }

  // ---- normalize and store context in (L,N,E) bf16 layout for out_proj
  float inv[8];
#pragma unroll
  for (int r = 0; r < 8; ++r) inv[r] = 1.0f / sl[r];
  const int n = bh >> 3, h = bh & 7;
#pragma unroll
  for (int nt4 = 0; nt4 < 4; ++nt4) {
    int d = nt4 * 16 + (lane & 15);
#pragma unroll
    for (int r = 0; r < 8; ++r) {
      int l = l0 + r + half * 8;
      ctx[((size_t)l * NB + n) * EMB + h * HD + d] = f2bf(acc[nt4][r] * inv[r]);
    }
  }
}

// ---------------------------------------------------------------------------
// Kernel 3: attn.sum(axis=1)/H, deterministic (one wave per (head, key tile),
// loops all 128 query tiles recomputing QKᵀ with saved max/denominator).
// ---------------------------------------------------------------------------
__global__ void __launch_bounds__(32)
k_colsum(const uint16_t* __restrict__ qb, const uint16_t* __restrict__ kb,
         const float* __restrict__ rowmax, const float* __restrict__ denom,
         float* __restrict__ attnsum) {
  const int lane = threadIdx.x & 31;
  const int half = (lane >> 4) & 1;
  const int st = blockIdx.x;                   // 0..128 key tiles
  const int bh = blockIdx.y;
  const uint16_t* kbase = kb + (size_t)bh * SPAD * HD + (size_t)st * 16 * HD;
  const v16bf bk0 = b_frag_nk_bf(kbase, HD, 0,  lane);
  const v16bf bk1 = b_frag_nk_bf(kbase, HD, 32, lane);
  float colacc = 0.f;
  for (int lt = 0; lt < 128; ++lt) {
    const uint16_t* qbase = qb + ((size_t)bh * LSEQ + (size_t)lt * 16) * HD;
    __builtin_prefetch(qbase + (size_t)16 * HD, 0, 0);
    v8f c = {};
    c = wmma_bf16(a_frag_bf(qbase,      HD, lane), bk0, c);
    c = wmma_bf16(a_frag_bf(qbase + 32, HD, lane), bk1, c);
#pragma unroll
    for (int r = 0; r < 8; ++r) {
      int row = lt * 16 + r + half * 8;
      float mr = rowmax[(size_t)bh * LSEQ + row];
      float dr = denom [(size_t)bh * LSEQ + row];
      colacc += __expf(c[r] - mr) / dr;
    }
  }
  colacc += __shfl_xor(colacc, 16, 32);        // combine row halves
  const int sg = st * 16 + (lane & 15);
  if (lane < 16 && sg < SEQS)
    attnsum[(size_t)bh * SEQS + sg] = colacc * 0.125f;   // / H
}

// ---------------------------------------------------------------------------
// Kernel 4: out_proj GEMM: ctx(8192x512) @ Woᵀ(512x512) + bias -> f32 out.
// ---------------------------------------------------------------------------
__global__ void __launch_bounds__(128)
k_outproj(const uint16_t* __restrict__ ctx, const uint16_t* __restrict__ W,
          const float* __restrict__ bias, float* __restrict__ out) {
  const int lane = threadIdx.x & 31;
  const int wave = threadIdx.x >> 5;
  const int mt = blockIdx.x;                   // 0..511
  const int nt = blockIdx.y * 4 + wave;        // 0..31
  const int half = (lane >> 4) & 1;
  const int col = nt * 16 + (lane & 15);
  v8f c = {};
#pragma unroll 4
  for (int kc = 0; kc < EMB; kc += 32) {
    __builtin_prefetch(ctx + (size_t)mt * 16 * EMB + kc + 128, 0, 0);
    v16bf a = a_frag_bf(ctx + (size_t)mt * 16 * EMB + kc, EMB, lane);
    v16bf b = b_frag_nk_bf(W + (size_t)nt * 16 * EMB, EMB, kc, lane);
    c = wmma_bf16(a, b, c);
  }
  const float bc = bias[col];
#pragma unroll
  for (int r = 0; r < 8; ++r) {
    int rg = mt * 16 + r + half * 8;
    out[(size_t)rg * EMB + col] = c[r] + bc;
  }
}

// ---------------------------------------------------------------------------
extern "C" void kernel_launch(void* const* d_in, const int* in_sizes, int n_in,
                              void* d_out, int out_size, void* d_ws, size_t ws_size,
                              hipStream_t stream) {
  (void)in_sizes; (void)n_in; (void)out_size; (void)ws_size;
  const float* query  = (const float*)d_in[0];
  // d_in[1] (key) and d_in[2] (value) are ignored, faithful to the reference.
  const float* Win    = (const float*)d_in[3];
  const float* bin    = (const float*)d_in[4];
  const float* Wout   = (const float*)d_in[5];
  const float* bout   = (const float*)d_in[6];
  const float* bias_k = (const float*)d_in[7];
  const float* bias_v = (const float*)d_in[8];

  // workspace layout (bf16 buffers first, then f32 stats) ~44.8 MB total
  uint16_t* xbf   = (uint16_t*)d_ws;                     // 8192*512
  uint16_t* wibf  = xbf  + (size_t)LSEQ * NB * EMB;      // 1536*512
  uint16_t* wobf  = wibf + (size_t)E3 * EMB;             // 512*512
  uint16_t* qb    = wobf + (size_t)EMB * EMB;            // 32*2048*64
  uint16_t* kb    = qb   + (size_t)BH * LSEQ * HD;       // 32*2080*64
  uint16_t* vt    = kb   + (size_t)BH * SPAD * HD;       // 32*64*2080 (d-major)
  uint16_t* ctx   = vt   + (size_t)BH * HD * SPAD;       // 8192*512
  float* rowmax = (float*)(ctx + (size_t)BH * LSEQ * HD);
  float* denomv = rowmax + (size_t)BH * LSEQ;

  float* out     = (float*)d_out;                        // (L,N,E)
  float* attnsum = out + (size_t)LSEQ * NB * EMB;        // (BH, S)

  const int nX = LSEQ * NB * EMB, nWi = E3 * EMB, nWo = EMB * EMB;
  k_cvt    <<<dim3(nX  / 256), 256, 0, stream>>>(query, xbf, nX);
  k_cvt    <<<dim3(nWi / 256), 256, 0, stream>>>(Win,  wibf, nWi);
  k_cvt    <<<dim3(nWo / 256), 256, 0, stream>>>(Wout, wobf, nWo);
  k_init   <<<dim3(256),      256, 0, stream>>>(bias_k, bias_v, kb, vt);
  k_qkv    <<<dim3(512, 24),  128, 0, stream>>>(xbf, wibf, bin, qb, kb, vt);
  k_attn   <<<dim3(128, 32),   32, 0, stream>>>(qb, kb, vt, ctx, rowmax, denomv);
  k_colsum <<<dim3(129, 32),   32, 0, stream>>>(qb, kb, rowmax, denomv, attnsum);
  k_outproj<<<dim3(512, 8),   128, 0, stream>>>(ctx, wobf, bout, out);
}
